// ViT_32710470926452
// MI455X (gfx1250) — compile-verified, matrix operand order checked by reading, not run on hardware
//
#include <hip/hip_runtime.h>
#include <cstdint>
#include <cstddef>

// ---------------- model constants ----------------
#define Dm    768
#define Lm    12
#define NHm   12
#define HDm   64
#define MLPm  3072
#define Bm    8
#define Cm    3
#define Hm    384
#define Pm    16
#define GHm   24
#define GWm   24
#define NPATCH 576
#define NTOK   577           // 576 patches + CLS
#define NPAD   608           // padded to multiple of 32 (19*32)
#define ROWS   (Bm * NPAD)   // 4864
#define NCLSm  1000

typedef _Float16 h16;
typedef __attribute__((ext_vector_type(8)))  _Float16 v8h;
typedef __attribute__((ext_vector_type(16))) _Float16 v16h;
typedef __attribute__((ext_vector_type(8)))  float    v8f;

// ---------------- WMMA GEMM ----------------
// C[m][n] = sum_k A[m][k] * B(k,n)  (+bias, +gelu, +residual)
// BT=true : B stored as Bt[n][k]; 128x128 block, 32x64 wave tile, 8 WMMA/K-step
// BT=false: B stored as B[k][n] ; 128x64  block, 32x32 wave tile, 4 WMMA/K-step
struct GemmP {
  const h16* A; const h16* B; const float* bias;
  float* C32; h16* C16;
  int M, N, K, lda, ldb, ldc, zdiv, flags;   // flags: 1=bias, 2=add-residual(C32), 4=gelu
  long long sAo, sAi, sBo, sBi, sCo, sCi;    // off = (z/zdiv)*so + (z%zdiv)*si
};

__device__ __forceinline__ float gelu_exact(float v) {
  return 0.5f * v * (1.0f + erff(v * 0.70710678118654752f));
}

template <bool BT>
__global__ __launch_bounds__(256) void gemm_wmma(GemmP p) {
  constexpr int BM = 128;
  constexpr int BN = BT ? 128 : 64;
  constexpr int BK = 32;
  constexpr int WN = BT ? 64 : 32;       // wave tile columns (2 waves across)
  constexpr int NACC = WN / 16;          // 4 (NT) or 2 (NN) accumulators per row-slab
  constexpr int ASTR = BK + 8;           // 40 halves -> 80B rows (16B aligned)
  constexpr int BSTR = BT ? (BK + 8) : (BN + 8);
  __shared__ alignas(16) h16 As[BM * ASTR];
  __shared__ alignas(16) h16 Bs[BT ? (BN * (BK + 8)) : (BK * (BN + 8))];

  const int z = blockIdx.z;
  const long long offA = (long long)(z / p.zdiv) * p.sAo + (long long)(z % p.zdiv) * p.sAi;
  const long long offB = (long long)(z / p.zdiv) * p.sBo + (long long)(z % p.zdiv) * p.sBi;
  const long long offC = (long long)(z / p.zdiv) * p.sCo + (long long)(z % p.zdiv) * p.sCi;
  const h16* __restrict__ Ag = p.A + offA;
  const h16* __restrict__ Bg = p.B + offB;

  const int m0 = blockIdx.y * BM;
  const int n0 = blockIdx.x * BN;
  const int tid  = threadIdx.x;
  const int lane = tid & 31;
  const int wave = tid >> 5;
  const int wrow = wave >> 1;            // 0..3 -> 32-row slab
  const int wcol = wave & 1;             // 0..1 -> WN-col slab

  const int row16 = lane & 15;
  const int kb    = (lane & 16) ? 8 : 0; // half-K select per ISA 16-bit A/B layout

  v8f acc[2][NACC];
#pragma unroll
  for (int s = 0; s < 2; ++s)
#pragma unroll
    for (int j = 0; j < NACC; ++j) acc[s][j] = v8f{};

  for (int k0 = 0; k0 < p.K; k0 += BK) {
    // ---- stage A tile: 128 rows x 32 halves (32B per thread) ----
    {
      const int ar = tid >> 1, ac = (tid & 1) * 16;
      const int gm = m0 + ar;
      v8h v0 = {}, v1 = {};
      if (gm < p.M) {
        const h16* src = Ag + (long long)gm * p.lda + k0 + ac;
        __builtin_prefetch(src + BK, 0, 3);     // global_prefetch of next K tile
        v0 = *(const v8h*)src;
        v1 = *(const v8h*)(src + 8);
      }
      *(v8h*)&As[ar * ASTR + ac]     = v0;
      *(v8h*)&As[ar * ASTR + ac + 8] = v1;
    }
    // ---- stage B tile ----
    if (BT) {
      const int br = tid >> 1, bc = (tid & 1) * 16;
      const int gn = n0 + br;
      v8h v0 = {}, v1 = {};
      if (gn < p.N) {
        const h16* src = Bg + (long long)gn * p.ldb + k0 + bc;
        v0 = *(const v8h*)src;
        v1 = *(const v8h*)(src + 8);
      }
      *(v8h*)&Bs[br * BSTR + bc]     = v0;
      *(v8h*)&Bs[br * BSTR + bc + 8] = v1;
    } else {
      const int kr = tid >> 3, nc = (tid & 7) * 8;
      const h16* src = Bg + (long long)(k0 + kr) * p.ldb + n0 + nc;
#pragma unroll
      for (int i = 0; i < 8; ++i) {
        h16 v = (h16)0.f;
        if (n0 + nc + i < p.N) v = src[i];
        Bs[kr * BSTR + nc + i] = v;
      }
    }
    __syncthreads();

    // ---- A fragments: two aligned 16B LDS reads each ----
    v16h afr[2];
#pragma unroll
    for (int s = 0; s < 2; ++s) {
      const h16* ap = &As[(wrow * 32 + s * 16 + row16) * ASTR + kb];
      v8h lo = *(const v8h*)ap;
      v8h hi = *(const v8h*)(ap + 16);
#pragma unroll
      for (int i = 0; i < 8; ++i) { afr[s][i] = lo[i]; afr[s][i + 8] = hi[i]; }
    }
    // ---- B fragments ----
    v16h bfr[NACC];
#pragma unroll
    for (int j = 0; j < NACC; ++j) {
      if (BT) {
        const h16* bp = &Bs[(wcol * WN + j * 16 + row16) * BSTR + kb];
        v8h lo = *(const v8h*)bp;
        v8h hi = *(const v8h*)(bp + 16);
#pragma unroll
        for (int i = 0; i < 8; ++i) { bfr[j][i] = lo[i]; bfr[j][i + 8] = hi[i]; }
      } else {
        const int ncol = wcol * WN + j * 16 + row16;
#pragma unroll
        for (int i = 0; i < 8; ++i) {
          bfr[j][i]     = Bs[(kb + i) * BSTR + ncol];
          bfr[j][i + 8] = Bs[(16 + kb + i) * BSTR + ncol];
        }
      }
    }
    // ---- 8 (NT) / 4 (NN) WMMAs per K-step ----
#pragma unroll
    for (int s = 0; s < 2; ++s)
#pragma unroll
      for (int j = 0; j < NACC; ++j)
        acc[s][j] = __builtin_amdgcn_wmma_f32_16x16x32_f16(
            false, afr[s], false, bfr[j], (short)0, acc[s][j], false, false);
    __syncthreads();
  }

  // ---- epilogue (C/D layout: VGPR r -> M = r + 8*(lane>=16), N = lane&15) ----
#pragma unroll
  for (int s = 0; s < 2; ++s) {
    const int mbase = m0 + wrow * 32 + s * 16 + ((lane >> 4) << 3);
#pragma unroll
    for (int j = 0; j < NACC; ++j) {
      const int n = n0 + wcol * WN + j * 16 + (lane & 15);
      if (n >= p.N) continue;
      const float bv = (p.flags & 1) ? p.bias[n] : 0.0f;
#pragma unroll
      for (int r = 0; r < 8; ++r) {
        const int m = mbase + r;
        if (m >= p.M) continue;
        const long long ci = offC + (long long)m * p.ldc + n;
        float v = acc[s][j][r] + bv;
        if (p.flags & 4) v = gelu_exact(v);
        if (p.flags & 2) v += p.C32[ci];
        if (p.C32) p.C32[ci] = v;
        if (p.C16) p.C16[ci] = (h16)v;
      }
    }
  }
}

// ---------------- helper kernels ----------------

// faithful replication of the reference's flatten order:
// flat index over (p1, b, c, gh, gw, p2) equals dest flat of [B, 576, 768]
__global__ __launch_bounds__(256) void patch_gather_k(const float* __restrict__ x,
                                                      h16* __restrict__ patches,
                                                      int total) {
  int tid = blockIdx.x * blockDim.x + threadIdx.x;
  if (tid >= total) return;
  int t  = tid;
  int p2 = t & 15;        t >>= 4;
  int gw = t % GWm;       t /= GWm;
  int gh = t % GHm;       t /= GHm;
  int c  = t % Cm;        t /= Cm;
  int b  = t & 7;         t >>= 3;
  int p1 = t;             // 0..15
  long long src = (((long long)(b * Cm + c) * Hm + (gh * Pm + p1)) * (long long)Hm)
                  + (gw * Pm + p2);
  patches[tid] = (h16)x[src];
}

// fp32 [R][C] -> f16 [C][R]
__global__ __launch_bounds__(256) void transpose_f16_k(const float* __restrict__ in,
                                                       h16* __restrict__ out,
                                                       int R, int C) {
  int tid = blockIdx.x * blockDim.x + threadIdx.x;
  if (tid >= R * C) return;
  int c = tid / R, r = tid % R;
  out[tid] = (h16)in[(long long)r * C + c];
}

__global__ __launch_bounds__(256) void convert_f16_k(const float* __restrict__ in,
                                                     h16* __restrict__ out, int n) {
  int tid = blockIdx.x * blockDim.x + threadIdx.x;
  if (tid < n) out[tid] = (h16)in[tid];
}

// h[b][0] = cls + pos[0]; h[b][1..576] += pos[n]; h[b][577..607] = 0
__global__ __launch_bounds__(256) void finalize_embed_k(float* __restrict__ h,
                                                        const float* __restrict__ cls,
                                                        const float* __restrict__ pos) {
  long long tid = (long long)blockIdx.x * blockDim.x + threadIdx.x;
  if (tid >= (long long)ROWS * Dm) return;
  int d = (int)(tid % Dm);
  int r = (int)((tid / Dm) % NPAD);
  if (r == 0)          h[tid] = cls[d] + pos[d];
  else if (r < NTOK)   h[tid] += pos[(long long)r * Dm + d];
  else                 h[tid] = 0.0f;
}

// LayerNorm over D=768, output f16; one block per row
__global__ __launch_bounds__(256) void layernorm_k(const float* __restrict__ x,
                                                   const float* __restrict__ g,
                                                   const float* __restrict__ b,
                                                   h16* __restrict__ y) {
  const long long row = blockIdx.x;
  const float* xr = x + row * Dm;
  float s = 0.f, s2 = 0.f;
  for (int i = threadIdx.x; i < Dm; i += 256) { float v = xr[i]; s += v; s2 += v * v; }
  __shared__ float rs[256], rq[256];
  rs[threadIdx.x] = s; rq[threadIdx.x] = s2;
  __syncthreads();
  for (int off = 128; off > 0; off >>= 1) {
    if (threadIdx.x < off) { rs[threadIdx.x] += rs[threadIdx.x + off];
                             rq[threadIdx.x] += rq[threadIdx.x + off]; }
    __syncthreads();
  }
  const float mean = rs[0] * (1.0f / Dm);
  float var = rq[0] * (1.0f / Dm) - mean * mean;
  var = fmaxf(var, 0.0f);
  const float inv = rsqrtf(var + 1e-5f);
  for (int i = threadIdx.x; i < Dm; i += 256)
    y[row * Dm + i] = (h16)((xr[i] - mean) * inv * g[i] + b[i]);
}

// masked, scaled softmax in-place on f16 scores [96][608][608]
__global__ __launch_bounds__(256) void softmax_k(h16* __restrict__ S) {
  const long long row = blockIdx.x;
  h16* sr = S + row * NPAD;
  float v[3];
  float mx = -1e30f;
  int idx = 0;
  for (int c = threadIdx.x; c < NPAD; c += 256, ++idx) {
    float f = (c < NTOK) ? (float)sr[c] * 0.125f : -1e30f;  // 1/sqrt(64)
    v[idx] = f; mx = fmaxf(mx, f);
  }
  __shared__ float red[256];
  red[threadIdx.x] = mx; __syncthreads();
  for (int off = 128; off > 0; off >>= 1) {
    if (threadIdx.x < off) red[threadIdx.x] = fmaxf(red[threadIdx.x], red[threadIdx.x + off]);
    __syncthreads();
  }
  mx = red[0]; __syncthreads();
  float sum = 0.f; idx = 0;
  for (int c = threadIdx.x; c < NPAD; c += 256, ++idx) {
    float e = __expf(v[idx] - mx);
    v[idx] = e; sum += e;
  }
  red[threadIdx.x] = sum; __syncthreads();
  for (int off = 128; off > 0; off >>= 1) {
    if (threadIdx.x < off) red[threadIdx.x] += red[threadIdx.x + off];
    __syncthreads();
  }
  const float inv = 1.0f / red[0];
  idx = 0;
  for (int c = threadIdx.x; c < NPAD; c += 256, ++idx)
    sr[c] = (h16)(v[idx] * inv);
}

// ---------------- host launcher ----------------
extern "C" void kernel_launch(void* const* d_in, const int* in_sizes, int n_in,
                              void* d_out, int out_size, void* d_ws, size_t ws_size,
                              hipStream_t stream) {
  (void)in_sizes; (void)n_in; (void)out_size; (void)ws_size;
  const float* x      = (const float*)d_in[0];
  const float* proj_w = (const float*)d_in[1];
  const float* proj_b = (const float*)d_in[2];
  const float* cls_e  = (const float*)d_in[3];
  const float* pos_e  = (const float*)d_in[4];
  const float* ln1_g  = (const float*)d_in[5];
  const float* ln1_b  = (const float*)d_in[6];
  const float* qw     = (const float*)d_in[7];
  const float* qb     = (const float*)d_in[8];
  const float* kw     = (const float*)d_in[9];
  const float* kb     = (const float*)d_in[10];
  const float* vw     = (const float*)d_in[11];
  const float* vb     = (const float*)d_in[12];
  const float* ow     = (const float*)d_in[13];
  const float* ob     = (const float*)d_in[14];
  const float* ln2_g  = (const float*)d_in[15];
  const float* ln2_b  = (const float*)d_in[16];
  const float* fcw    = (const float*)d_in[17];
  const float* fcb    = (const float*)d_in[18];
  const float* pww    = (const float*)d_in[19];
  const float* pwb    = (const float*)d_in[20];
  const float* lnf_g  = (const float*)d_in[21];
  const float* lnf_b  = (const float*)d_in[22];
  const float* head_w = (const float*)d_in[23];
  const float* head_b = (const float*)d_in[24];
  float* out = (float*)d_out;

  // ---- workspace carve-up ----
  char* wsp = (char*)d_ws;
  auto carve = [&](size_t bytes) -> char* {
    char* p = wsp;
    wsp += (bytes + 255) & ~(size_t)255;
    return p;
  };
  h16*   patches = (h16*)  carve((size_t)Bm * NPATCH * Dm * 2);
  float* h       = (float*)carve((size_t)ROWS * Dm * 4);
  h16*   ybuf    = (h16*)  carve((size_t)ROWS * Dm * 2);
  h16*   qbuf    = (h16*)  carve((size_t)ROWS * Dm * 2);
  h16*   kbuf    = (h16*)  carve((size_t)ROWS * Dm * 2);
  h16*   vbuf    = (h16*)  carve((size_t)ROWS * Dm * 2);
  h16*   attnb   = (h16*)  carve((size_t)ROWS * Dm * 2);
  h16*   scores  = (h16*)  carve((size_t)Bm * NHm * NPAD * NPAD * 2);
  h16*   mbuf    = (h16*)  carve((size_t)ROWS * MLPm * 2);
  h16*   wq      = (h16*)  carve((size_t)Dm * Dm * 2);
  h16*   wk      = (h16*)  carve((size_t)Dm * Dm * 2);
  h16*   wv      = (h16*)  carve((size_t)Dm * Dm * 2);
  h16*   wo      = (h16*)  carve((size_t)Dm * Dm * 2);
  h16*   wfc     = (h16*)  carve((size_t)Dm * MLPm * 2);
  h16*   wpw     = (h16*)  carve((size_t)Dm * MLPm * 2);
  h16*   wproj   = (h16*)  carve((size_t)Dm * Dm * 2);
  h16*   whead   = (h16*)  carve((size_t)Dm * NCLSm * 2);

  auto gemm = [&](bool bt, const h16* A, const h16* Bp, const float* bias,
                  float* C32, h16* C16, int M, int N, int K,
                  int lda, int ldb, int ldc, int Z, int zdiv,
                  long long sAo, long long sAi, long long sBo, long long sBi,
                  long long sCo, long long sCi, int flags) {
    GemmP p{A, Bp, bias, C32, C16, M, N, K, lda, ldb, ldc, zdiv, flags,
            sAo, sAi, sBo, sBi, sCo, sCi};
    const int bn = bt ? 128 : 64;
    dim3 grid((N + bn - 1) / bn, (M + 127) / 128, Z);
    if (bt) gemm_wmma<true><<<grid, 256, 0, stream>>>(p);
    else    gemm_wmma<false><<<grid, 256, 0, stream>>>(p);
  };

  // ---- patch embed ----
  {
    const int total = Bm * NPATCH * Dm;
    patch_gather_k<<<(total + 255) / 256, 256, 0, stream>>>(x, patches, total);
    transpose_f16_k<<<(Dm * Dm + 255) / 256, 256, 0, stream>>>(proj_w, wproj, Dm, Dm);
    // emb = patches @ proj_w + proj_b -> rows (b*608 + 1 + np) of h
    gemm(true, patches, wproj, proj_b, h + Dm, nullptr,
         NPATCH, Dm, Dm, Dm, Dm, Dm, Bm, 1,
         (long long)NPATCH * Dm, 0, 0, 0, (long long)NPAD * Dm, 0, /*bias*/1);
    finalize_embed_k<<<((long long)ROWS * Dm + 255) / 256, 256, 0, stream>>>(h, cls_e, pos_e);
  }

  // ---- transformer layers ----
  const long long DD = (long long)Dm * Dm;
  for (int l = 0; l < Lm; ++l) {
    transpose_f16_k<<<(Dm * Dm + 255) / 256, 256, 0, stream>>>(qw + l * DD, wq, Dm, Dm);
    transpose_f16_k<<<(Dm * Dm + 255) / 256, 256, 0, stream>>>(kw + l * DD, wk, Dm, Dm);
    transpose_f16_k<<<(Dm * Dm + 255) / 256, 256, 0, stream>>>(vw + l * DD, wv, Dm, Dm);
    transpose_f16_k<<<(Dm * Dm + 255) / 256, 256, 0, stream>>>(ow + l * DD, wo, Dm, Dm);
    transpose_f16_k<<<(Dm * MLPm + 255) / 256, 256, 0, stream>>>(
        fcw + (long long)l * Dm * MLPm, wfc, Dm, MLPm);     // -> [3072][768]
    transpose_f16_k<<<(Dm * MLPm + 255) / 256, 256, 0, stream>>>(
        pww + (long long)l * Dm * MLPm, wpw, MLPm, Dm);     // -> [768][3072]

    // y = LN1(h)
    layernorm_k<<<ROWS, 256, 0, stream>>>(h, ln1_g + l * Dm, ln1_b + l * Dm, ybuf);

    // q,k,v = y @ W + b   (f16 outputs)
    gemm(true, ybuf, wq, qb + l * Dm, nullptr, qbuf, ROWS, Dm, Dm, Dm, Dm, Dm,
         1, 1, 0, 0, 0, 0, 0, 0, 1);
    gemm(true, ybuf, wk, kb + l * Dm, nullptr, kbuf, ROWS, Dm, Dm, Dm, Dm, Dm,
         1, 1, 0, 0, 0, 0, 0, 0, 1);
    gemm(true, ybuf, wv, vb + l * Dm, nullptr, vbuf, ROWS, Dm, Dm, Dm, Dm, Dm,
         1, 1, 0, 0, 0, 0, 0, 0, 1);

    // S = Q @ K^T  per (b,h): z = b*12 + h
    gemm(true, qbuf, kbuf, nullptr, nullptr, scores, NPAD, NPAD, HDm,
         Dm, Dm, NPAD, Bm * NHm, NHm,
         (long long)NPAD * Dm, HDm, (long long)NPAD * Dm, HDm,
         (long long)NHm * NPAD * NPAD, (long long)NPAD * NPAD, 0);

    // softmax (scale + column mask) in place
    softmax_k<<<Bm * NHm * NPAD, 256, 0, stream>>>(scores);

    // attn_out = P @ V  (NN path: V rows are the K dimension)
    gemm(false, scores, vbuf, nullptr, nullptr, attnb, NPAD, HDm, NPAD,
         NPAD, Dm, Dm, Bm * NHm, NHm,
         (long long)NHm * NPAD * NPAD, (long long)NPAD * NPAD,
         (long long)NPAD * Dm, HDm, (long long)NPAD * Dm, HDm, 0);

    // h += attn_out @ oW + oB
    gemm(true, attnb, wo, ob + l * Dm, h, nullptr, ROWS, Dm, Dm, Dm, Dm, Dm,
         1, 1, 0, 0, 0, 0, 0, 0, 1 | 2);

    // y2 = LN2(h)
    layernorm_k<<<ROWS, 256, 0, stream>>>(h, ln2_g + l * Dm, ln2_b + l * Dm, ybuf);

    // m = gelu(y2 @ fcW + fcB)   (f16)
    gemm(true, ybuf, wfc, fcb + (long long)l * MLPm, nullptr, mbuf,
         ROWS, MLPm, Dm, Dm, Dm, MLPm, 1, 1, 0, 0, 0, 0, 0, 0, 1 | 4);

    // h += m @ pW + pB
    gemm(true, mbuf, wpw, pwb + l * Dm, h, nullptr, ROWS, Dm, MLPm,
         MLPm, MLPm, Dm, 1, 1, 0, 0, 0, 0, 0, 0, 1 | 2);
  }

  // ---- final LN + CLS head ----
  layernorm_k<<<ROWS, 256, 0, stream>>>(h, lnf_g, lnf_b, ybuf);
  convert_f16_k<<<(Dm * NCLSm + 255) / 256, 256, 0, stream>>>(head_w, whead, Dm * NCLSm);
  // A rows = CLS token of each batch via lda = NPAD*Dm; B = head_w [768][1000] (NN)
  gemm(false, ybuf, whead, head_b, out, nullptr, Bm, NCLSm, Dm,
       NPAD * Dm, NCLSm, NCLSm, 1, 1, 0, 0, 0, 0, 0, 0, 1);
}
